// MLSTM_17368847745629
// MI455X (gfx1250) — compile-verified, matrix operand order checked by reading, not run on hardware
//
#include <hip/hip_runtime.h>

#define I_DIMC 2048
#define H_DIMC 2048
#define G_DIMC 8192   // 4*H
#define SEQLC  4096
#define OUTDC  2048

typedef __attribute__((ext_vector_type(16))) __bf16 v16bf;
typedef __attribute__((ext_vector_type(8)))  __bf16 v8bf;
typedef __attribute__((ext_vector_type(8)))  float  v8f;

__device__ __forceinline__ unsigned short f2bf(float f) {
    unsigned u = __float_as_uint(f);
    u += 0x7FFFu + ((u >> 16) & 1u);   // round-to-nearest-even
    return (unsigned short)(u >> 16);
}

// 16-bit WMMA A/B fragment per ISA layout: lane (m = L&15, hs = L>>4) holds
// K = kb + hs*8 .. +8  (VGPR0-3) and K = kb + 16 + hs*8 .. +8 (VGPR4-7),
// i.e. two contiguous 16-byte chunks of a K-contiguous row.
__device__ __forceinline__ v16bf load_frag(const __bf16* __restrict__ row, int kb, int hs) {
    const v8bf* p0 = reinterpret_cast<const v8bf*>(row + kb + hs * 8);
    const v8bf* p1 = reinterpret_cast<const v8bf*>(row + kb + 16 + hs * 8);
    v8bf lo = *p0;
    v8bf hi = *p1;
    return __builtin_shufflevector(lo, hi, 0,1,2,3,4,5,6,7,8,9,10,11,12,13,14,15);
}

__device__ __forceinline__ v8f vzero8() {
    v8f z;
#pragma unroll
    for (int i = 0; i < 8; ++i) z[i] = 0.0f;
    return z;
}

#define WMMA_BF16(A, B, C) \
    __builtin_amdgcn_wmma_f32_16x16x32_bf16(false, (A), false, (B), (short)0, (C), false, false)

// ---------------- fp32 -> bf16 conversion ----------------
__global__ void __launch_bounds__(256) cvt_bf16(const float* __restrict__ src,
                                                unsigned short* __restrict__ dst, int n) {
    for (int i = blockIdx.x * 256 + threadIdx.x; i < n; i += gridDim.x * 256)
        dst[i] = f2bf(src[i]);
}

// ---------------- state / sync init (every call: determinism) ----------------
__global__ void __launch_bounds__(256) init_state(unsigned short* __restrict__ hbf,
                                                  float* __restrict__ hf,
                                                  unsigned* __restrict__ sync) {
    int i = blockIdx.x * 256 + threadIdx.x;
    if (i < 2 * H_DIMC) hbf[i] = 0;
    if (i < H_DIMC) hf[i] = 0.0f;
    if (i < 8) sync[i] = 0u;
}

// ---------------- xg = x @ W_ih^T + b_ih + b_hh  (bf16 WMMA GEMM) ----------------
// wave tile: 16 (seq) x 64 (gate); K loop 2048 in steps of 32, software-pipelined
// so next-iteration fragment loads overlap current-iteration WMMAs.
__global__ void __launch_bounds__(256) xg_gemm(const __bf16* __restrict__ Xb,
                                               const __bf16* __restrict__ Wb,
                                               const float* __restrict__ bih,
                                               const float* __restrict__ bhh,
                                               float* __restrict__ XG) {
    const int lane = threadIdx.x & 31;
    const int wv   = threadIdx.x >> 5;
    const int wid  = blockIdx.x * 8 + wv;
    const int NT   = G_DIMC / 64;          // 128 n-tiles
    const int mt   = wid / NT;             // 0..255
    const int nt   = wid % NT;
    const int m    = lane & 15;
    const int hs   = lane >> 4;

    const __bf16* arow = Xb + (size_t)(mt * 16 + m) * I_DIMC;
    const __bf16* b0r  = Wb + (size_t)(nt * 64 +  0 + m) * I_DIMC;
    const __bf16* b1r  = Wb + (size_t)(nt * 64 + 16 + m) * I_DIMC;
    const __bf16* b2r  = Wb + (size_t)(nt * 64 + 32 + m) * I_DIMC;
    const __bf16* b3r  = Wb + (size_t)(nt * 64 + 48 + m) * I_DIMC;

    v8f c0 = vzero8(), c1 = vzero8(), c2 = vzero8(), c3 = vzero8();

    // pipeline stage 0
    v16bf a  = load_frag(arow, 0, hs);
    v16bf b0 = load_frag(b0r,  0, hs);
    v16bf b1 = load_frag(b1r,  0, hs);
    v16bf b2 = load_frag(b2r,  0, hs);
    v16bf b3 = load_frag(b3r,  0, hs);

    for (int kb = 0; kb < I_DIMC - 32; kb += 32) {
        // issue next-stage loads before consuming current fragments
        v16bf an  = load_frag(arow, kb + 32, hs);
        v16bf bn0 = load_frag(b0r,  kb + 32, hs);
        v16bf bn1 = load_frag(b1r,  kb + 32, hs);
        v16bf bn2 = load_frag(b2r,  kb + 32, hs);
        v16bf bn3 = load_frag(b3r,  kb + 32, hs);
        c0 = WMMA_BF16(a, b0, c0);
        c1 = WMMA_BF16(a, b1, c1);
        c2 = WMMA_BF16(a, b2, c2);
        c3 = WMMA_BF16(a, b3, c3);
        a = an; b0 = bn0; b1 = bn1; b2 = bn2; b3 = bn3;
    }
    c0 = WMMA_BF16(a, b0, c0);
    c1 = WMMA_BF16(a, b1, c1);
    c2 = WMMA_BF16(a, b2, c2);
    c3 = WMMA_BF16(a, b3, c3);

#pragma unroll
    for (int sub = 0; sub < 4; ++sub) {
        v8f acc = (sub == 0) ? c0 : (sub == 1) ? c1 : (sub == 2) ? c2 : c3;
        const int col = nt * 64 + sub * 16 + m;        // D: N = lane&15
        const float bias = bih[col] + bhh[col];
#pragma unroll
        for (int r = 0; r < 8; ++r) {                  // D: M = hs*8 + r
            const int row = mt * 16 + hs * 8 + r;
            XG[(size_t)row * G_DIMC + col] = acc[r] + bias;
        }
    }
}

// ---------------- sequential LSTM recurrence (persistent, grid-synced) ----------------
// 128 blocks x 256 threads. Block b owns h elements [16b, 16b+16).
// 8 waves split K=2048 (256 each); each wave computes partials of all 4 gates
// (A-frag = broadcast h, one A load feeds 4 WMMAs), software-pipelined K loop.
// LDS reduce, 16 threads run the cell; c stays in registers; h double-buffered.
__global__ void __launch_bounds__(256) lstm_seq(const __bf16* __restrict__ Whb,
                                                const float* __restrict__ XG,
                                                unsigned short* __restrict__ hbf, // 2*H
                                                float* __restrict__ hf,
                                                unsigned* __restrict__ sync) {
    __shared__ float part[8][4][16];
    const int lane = threadIdx.x & 31;
    const int wv   = threadIdx.x >> 5;
    const int m    = lane & 15;
    const int hs   = lane >> 4;
    const int e0   = blockIdx.x * 16;
    const int kbase = wv * 256;
    const unsigned nblk = gridDim.x;

    const __bf16* br0 = Whb + (size_t)(0 * H_DIMC + e0 + m) * H_DIMC;
    const __bf16* br1 = Whb + (size_t)(1 * H_DIMC + e0 + m) * H_DIMC;
    const __bf16* br2 = Whb + (size_t)(2 * H_DIMC + e0 + m) * H_DIMC;
    const __bf16* br3 = Whb + (size_t)(3 * H_DIMC + e0 + m) * H_DIMC;

    float cstate = 0.0f;                 // owned by threadIdx.x < 16
    for (int t = 0; t < SEQLC; ++t) {
        const __bf16* hsrc = reinterpret_cast<const __bf16*>(hbf) + (size_t)(t & 1) * H_DIMC;

        // hoist xg[t] loads so they fly during the WMMA loop
        float x0 = 0.f, x1 = 0.f, x2 = 0.f, x3 = 0.f;
        if (threadIdx.x < 16) {
            const float* xgt = XG + (size_t)t * G_DIMC + e0 + threadIdx.x;
            x0 = xgt[0 * H_DIMC];
            x1 = xgt[1 * H_DIMC];
            x2 = xgt[2 * H_DIMC];
            x3 = xgt[3 * H_DIMC];
        }

        v8f a0 = vzero8(), a1 = vzero8(), a2 = vzero8(), a3 = vzero8();

        v16bf af = load_frag(hsrc, kbase, hs);
        v16bf w0 = load_frag(br0, kbase, hs);
        v16bf w1 = load_frag(br1, kbase, hs);
        v16bf w2 = load_frag(br2, kbase, hs);
        v16bf w3 = load_frag(br3, kbase, hs);
#pragma unroll
        for (int kk = 0; kk < 7; ++kk) {
            const int kb = kbase + kk * 32 + 32;
            v16bf afn = load_frag(hsrc, kb, hs);
            v16bf wn0 = load_frag(br0, kb, hs);
            v16bf wn1 = load_frag(br1, kb, hs);
            v16bf wn2 = load_frag(br2, kb, hs);
            v16bf wn3 = load_frag(br3, kb, hs);
            a0 = WMMA_BF16(af, w0, a0);
            a1 = WMMA_BF16(af, w1, a1);
            a2 = WMMA_BF16(af, w2, a2);
            a3 = WMMA_BF16(af, w3, a3);
            af = afn; w0 = wn0; w1 = wn1; w2 = wn2; w3 = wn3;
        }
        a0 = WMMA_BF16(af, w0, a0);
        a1 = WMMA_BF16(af, w1, a1);
        a2 = WMMA_BF16(af, w2, a2);
        a3 = WMMA_BF16(af, w3, a3);

        if (lane < 16) {                 // D row 0: VGPR0, lanes 0-15 = N=0..15
            part[wv][0][lane] = a0[0];
            part[wv][1][lane] = a1[0];
            part[wv][2][lane] = a2[0];
            part[wv][3][lane] = a3[0];
        }
        __syncthreads();
        if (threadIdx.x < 16) {
            const int n = threadIdx.x;
            float s0 = x0, s1 = x1, s2 = x2, s3 = x3;
#pragma unroll
            for (int w = 0; w < 8; ++w) {
                s0 += part[w][0][n]; s1 += part[w][1][n];
                s2 += part[w][2][n]; s3 += part[w][3][n];
            }
            const float ig = 1.0f / (1.0f + __expf(-s0));
            const float fg = 1.0f / (1.0f + __expf(-s1));
            const float gg = tanhf(s2);
            const float og = 1.0f / (1.0f + __expf(-s3));
            cstate = fg * cstate + ig * gg;
            const float hval = og * tanhf(cstate);
            hbf[(size_t)((t + 1) & 1) * H_DIMC + e0 + n] = f2bf(hval);
            if (t == SEQLC - 1) hf[e0 + n] = hval;
        }
        __threadfence();
        // ---- device-wide barrier (generation counter) ----
        __syncthreads();
        if (threadIdx.x == 0) {
            unsigned g = __hip_atomic_load(&sync[1], __ATOMIC_RELAXED, __HIP_MEMORY_SCOPE_AGENT);
            unsigned a = __hip_atomic_fetch_add(&sync[0], 1u, __ATOMIC_ACQ_REL, __HIP_MEMORY_SCOPE_AGENT);
            if (a == nblk - 1u) {
                __hip_atomic_store(&sync[0], 0u, __ATOMIC_RELAXED, __HIP_MEMORY_SCOPE_AGENT);
                __hip_atomic_fetch_add(&sync[1], 1u, __ATOMIC_ACQ_REL, __HIP_MEMORY_SCOPE_AGENT);
            } else {
                while (__hip_atomic_load(&sync[1], __ATOMIC_ACQUIRE, __HIP_MEMORY_SCOPE_AGENT) == g) {
                    __builtin_amdgcn_s_sleep(8);
                }
            }
        }
        __syncthreads();
    }
}

// ---------------- final FC: out = h_last @ W_fc^T + b_fc ----------------
__global__ void __launch_bounds__(256) fc_out(const float* __restrict__ hf,
                                              const float* __restrict__ Wfc,
                                              const float* __restrict__ bfc,
                                              float* __restrict__ out) {
    const int lane = threadIdx.x & 31;
    const int wv   = threadIdx.x >> 5;
    const int o    = blockIdx.x * 8 + wv;
    const float* wr = Wfc + (size_t)o * H_DIMC;
    float s = 0.0f;
    for (int k = lane; k < H_DIMC; k += 32) s += hf[k] * wr[k];
#pragma unroll
    for (int off = 16; off > 0; off >>= 1) s += __shfl_xor(s, off, 32);
    if (lane == 0) out[o] = s + bfc[o];
}

extern "C" void kernel_launch(void* const* d_in, const int* in_sizes, int n_in,
                              void* d_out, int out_size, void* d_ws, size_t ws_size,
                              hipStream_t stream) {
    (void)in_sizes; (void)n_in; (void)out_size; (void)ws_size;
    const float* x   = (const float*)d_in[0];   // [1, 4096, 2048]
    const float* Wih = (const float*)d_in[1];   // [8192, 2048]
    const float* Whh = (const float*)d_in[2];   // [8192, 2048]
    const float* bih = (const float*)d_in[3];   // [8192]
    const float* bhh = (const float*)d_in[4];   // [8192]
    const float* Wfc = (const float*)d_in[5];   // [2048, 2048]
    const float* bfc = (const float*)d_in[6];   // [2048]
    float* out = (float*)d_out;

    char* ws = (char*)d_ws;
    size_t off = 0;
    unsigned short* Wih_b = (unsigned short*)(ws + off); off += (size_t)G_DIMC * I_DIMC * 2;
    unsigned short* Whh_b = (unsigned short*)(ws + off); off += (size_t)G_DIMC * H_DIMC * 2;
    unsigned short* X_b   = (unsigned short*)(ws + off); off += (size_t)SEQLC * I_DIMC * 2;
    float*          XG    = (float*)(ws + off);          off += (size_t)SEQLC * G_DIMC * 4;
    unsigned short* hbf   = (unsigned short*)(ws + off); off += (size_t)2 * H_DIMC * 2;
    float*          hf    = (float*)(ws + off);          off += (size_t)H_DIMC * 4;
    unsigned*       sync  = (unsigned*)(ws + off);       off += 256;

    cvt_bf16<<<2048, 256, 0, stream>>>(Wih, Wih_b, G_DIMC * I_DIMC);
    cvt_bf16<<<2048, 256, 0, stream>>>(Whh, Whh_b, G_DIMC * H_DIMC);
    cvt_bf16<<<2048, 256, 0, stream>>>(x,   X_b,   SEQLC * I_DIMC);
    init_state<<<16, 256, 0, stream>>>(hbf, hf, sync);

    // xg GEMM: 256 m-tiles * 128 n-tiles = 32768 wave tiles / 8 waves per block
    xg_gemm<<<4096, 256, 0, stream>>>((const __bf16*)X_b, (const __bf16*)Wih_b, bih, bhh, XG);

    // persistent recurrence: 128 blocks (one per 16-element h slice), must co-reside
    lstm_seq<<<128, 256, 0, stream>>>((const __bf16*)Whh_b, XG, hbf, hf, sync);

    // final projection: one wave per output row
    fc_out<<<256, 256, 0, stream>>>(hf, Wfc, bfc, out);
}
